// SparseMOE_50646254354974
// MI455X (gfx1250) — matrix operand order repaired and checked by reading, not compiled, not run on hardware
//
#include <hip/hip_runtime.h>

#define TKN 2048
#define DM  1024
#define DH  2048
#define NE  8

typedef unsigned short u16;
typedef unsigned int   u32;
typedef __attribute__((ext_vector_type(16))) __bf16 v16bf;
typedef __attribute__((ext_vector_type(8)))  float  v8f;
typedef __attribute__((ext_vector_type(4)))  u32    v4u;

union FragBF { v16bf v; v4u q[2]; };

__device__ __forceinline__ u16 f2bf(float f) {
  u32 u = __builtin_bit_cast(u32, f);
  u += 0x7fffu + ((u >> 16) & 1u);   // round-to-nearest-even
  return (u16)(u >> 16);
}
__device__ __forceinline__ u32 pack2(float lo, float hi) {
  return (u32)f2bf(lo) | ((u32)f2bf(hi) << 16);
}
__device__ __forceinline__ void atomAddF32(float* p, float v) {
  __hip_atomic_fetch_add(p, v, __ATOMIC_RELAXED, __HIP_MEMORY_SCOPE_AGENT);
}
__device__ __forceinline__ v8f bfwmma(const FragBF& a, const FragBF& b, v8f c) {
  return __builtin_amdgcn_wmma_f32_16x16x32_bf16(false, a.v, false, b.v, (short)0, c, false, false);
}

// ------------------------------------------------ prep: convert x, zero state
__global__ __launch_bounds__(256) void moe_prep(
    const float* __restrict__ x, u16* __restrict__ xb,
    int* __restrict__ idxl, float* __restrict__ gatel,
    int* __restrict__ counts, float* __restrict__ psum, float* __restrict__ out)
{
  const long tid    = (long)blockIdx.x * blockDim.x + threadIdx.x;
  const long stride = (long)gridDim.x * blockDim.x;
  const long NX = (long)TKN * DM / 2;
  const float2* xf = (const float2*)x;
  for (long i = tid; i < NX; i += stride) { float2 f = xf[i]; ((u32*)xb)[i] = pack2(f.x, f.y); }
  for (long i = tid; i < (long)TKN * DM + 1; i += stride) out[i] = 0.0f;
  for (long i = tid; i < (long)NE * TKN; i += stride) { idxl[i] = 0; gatel[i] = 0.0f; }
  if (tid < NE) { counts[tid] = 0; psum[tid] = 0.0f; }
}

// ------------------------------------------------ tiled transpose fp32[RxC] -> bf16[CxR], per expert
__global__ __launch_bounds__(256) void moe_transpose_bf16(
    const float* __restrict__ src, u16* __restrict__ dst, int R, int C)
{
  __shared__ float tile[32][33];
  const int e = blockIdx.z;
  const long base = (long)e * R * C;
  const int r0 = blockIdx.y * 32;
  const int c0 = blockIdx.x * 32;
  const int col  = threadIdx.x & 31;
  const int row8 = threadIdx.x >> 5;     // 0..7
  #pragma unroll
  for (int j = 0; j < 4; ++j) {
    const int r = row8 + j * 8;
    tile[r][col] = src[base + (long)(r0 + r) * C + c0 + col];
  }
  __syncthreads();
  #pragma unroll
  for (int j = 0; j < 4; ++j) {
    const int r = row8 + j * 8;          // index along C in the output
    dst[base + (long)(c0 + r) * R + r0 + col] = f2bf(tile[col][r]);
  }
}

// ------------------------------------------------ router
__global__ __launch_bounds__(256) void moe_router(
    const float* __restrict__ x, const float* __restrict__ Wr,
    const float* __restrict__ br, int* __restrict__ idxl,
    float* __restrict__ gatel, int* __restrict__ counts, float* __restrict__ psum)
{
  const int t = blockIdx.x * 256 + threadIdx.x;
  if (t >= TKN) return;
  float acc[NE];
  #pragma unroll
  for (int e = 0; e < NE; ++e) acc[e] = br[e];
  const float* xr = x + (size_t)t * DM;
  for (int d = 0; d < DM; ++d) {
    const float xv = xr[d];
    #pragma unroll
    for (int e = 0; e < NE; ++e) acc[e] += xv * Wr[d * NE + e];
  }
  float mx = acc[0];
  #pragma unroll
  for (int e = 1; e < NE; ++e) mx = fmaxf(mx, acc[e]);
  float p[NE]; float s = 0.0f;
  #pragma unroll
  for (int e = 0; e < NE; ++e) { p[e] = __expf(acc[e] - mx); s += p[e]; }
  const float inv = 1.0f / s;
  #pragma unroll
  for (int e = 0; e < NE; ++e) { p[e] *= inv; atomAddF32(&psum[e], p[e]); }
  int i1 = 0;
  #pragma unroll
  for (int e = 1; e < NE; ++e) if (p[e] > p[i1]) i1 = e;
  int i2 = (i1 == 0) ? 1 : 0;
  #pragma unroll
  for (int e = 0; e < NE; ++e) if (e != i1 && p[e] > p[i2]) i2 = e;
  const int s1 = atomicAdd(&counts[i1], 1);
  idxl[i1 * TKN + s1] = t; gatel[i1 * TKN + s1] = p[i1];
  const int s2 = atomicAdd(&counts[i2], 1);
  idxl[i2 * TKN + s2] = t; gatel[i2 * TKN + s2] = p[i2];
}

// ------------------------------------------------ lb loss
__global__ void moe_loss(const float* __restrict__ psum, float* __restrict__ out_loss)
{
  if (threadIdx.x == 0 && blockIdx.x == 0) {
    float s = 0.0f;
    #pragma unroll
    for (int e = 0; e < NE; ++e) { float m = psum[e] / (float)TKN; s += m * m; }
    out_loss[0] = s * (float)NE;
  }
}

// ------------------------------------------------ FFN1: act = (xW1+b1) * silu(xW2+b2)
// Block tile M=32 x N=128, 8 waves (2x4), each wave: 2 n-subtiles, 4 WMMAs/step.
// Weights pre-transposed: W1t/W2t are [e][n(DH)][k(DM)] bf16.
__global__ __launch_bounds__(256) void moe_ffn1(
    const u16* __restrict__ xb, const u16* __restrict__ W1t, const u16* __restrict__ W2t,
    const float* __restrict__ b1, const float* __restrict__ b2,
    const int* __restrict__ counts, const int* __restrict__ idxl, u16* __restrict__ act)
{
  __shared__ u16 sA[2][32][40];     // [buf][m][k], 80B rows (16B aligned)
  __shared__ u16 sB1[2][128][40];   // [buf][n][k]
  __shared__ u16 sB2[2][128][40];

  const int e = blockIdx.z;
  const int cnt = counts[e];
  const int m_base = blockIdx.y * 32;
  if (m_base >= cnt) return;
  const int n_base = blockIdx.x * 128;
  const int tid  = threadIdx.x;
  const int lane = tid & 31;
  const int w    = tid >> 5;
  const int wm   = w >> 2;            // 0..1
  const int wn   = w & 3;             // 0..3
  const int hh   = lane >> 4;
  const int l16  = lane & 15;

  // A staging: threads 0..127, one v4u each: row=tid>>2 (0..31), k-off=(tid&3)*8
  const int arow = tid >> 2;
  const int ak   = (tid & 3) * 8;
  long tokbase = 0;
  if (tid < 128) tokbase = (long)idxl[e * TKN + m_base + arow] * DM;
  const u16* pA = xb + tokbase + ak;

  // B staging: all 256 threads, two v4u per matrix: rows brow and brow+64
  const int brow = tid >> 2;          // 0..63
  const int bko  = (tid & 3) * 8;
  const u16* pB1a = W1t + (long)e * DH * DM + (long)(n_base + brow) * DM + bko;
  const u16* pB1b = pB1a + (long)64 * DM;
  const u16* pB2a = W2t + (long)e * DH * DM + (long)(n_base + brow) * DM + bko;
  const u16* pB2b = pB2a + (long)64 * DM;

  // initial stage (k-slab 0) into buffer 0
  if (tid < 128) *(v4u*)&sA[0][arow][ak] = *(const v4u*)pA;
  *(v4u*)&sB1[0][brow][bko]      = *(const v4u*)pB1a;
  *(v4u*)&sB1[0][brow + 64][bko] = *(const v4u*)pB1b;
  *(v4u*)&sB2[0][brow][bko]      = *(const v4u*)pB2a;
  *(v4u*)&sB2[0][brow + 64][bko] = *(const v4u*)pB2b;
  __syncthreads();

  v8f hacc[2] = {{}, {}};
  v8f gacc[2] = {{}, {}};

  const int NT = DM / 32;
  for (int kt = 0; kt < NT; ++kt) {
    const int cur = kt & 1;
    const int kn  = (kt + 1) * 32;
    const bool more = (kt + 1 < NT);
    v4u ra = {}, r10 = {}, r11 = {}, r20 = {}, r21 = {};
    if (more) {                                   // stage next slab into regs
      if (tid < 128) ra = *(const v4u*)(pA + kn);
      r10 = *(const v4u*)(pB1a + kn);
      r11 = *(const v4u*)(pB1b + kn);
      r20 = *(const v4u*)(pB2a + kn);
      r21 = *(const v4u*)(pB2b + kn);
    }
    FragBF fa;
    fa.q[0] = *(const v4u*)&sA[cur][wm * 16 + l16][16 * hh];      // K 16h..16h+7
    fa.q[1] = *(const v4u*)&sA[cur][wm * 16 + l16][16 * hh + 8];  // K 16h+8..16h+15
    #pragma unroll
    for (int s = 0; s < 2; ++s) {
      const int nl = wn * 32 + s * 16 + l16;
      FragBF fb1, fb2;
      fb1.q[0] = *(const v4u*)&sB1[cur][nl][16 * hh];             // K 16h..16h+15
      fb1.q[1] = *(const v4u*)&sB1[cur][nl][16 * hh + 8];
      hacc[s] = bfwmma(fa, fb1, hacc[s]);
      fb2.q[0] = *(const v4u*)&sB2[cur][nl][16 * hh];
      fb2.q[1] = *(const v4u*)&sB2[cur][nl][16 * hh + 8];
      gacc[s] = bfwmma(fa, fb2, gacc[s]);
    }
    if (more) {
      const int nxt = cur ^ 1;
      if (tid < 128) *(v4u*)&sA[nxt][arow][ak] = ra;
      *(v4u*)&sB1[nxt][brow][bko]      = r10;
      *(v4u*)&sB1[nxt][brow + 64][bko] = r11;
      *(v4u*)&sB2[nxt][brow][bko]      = r20;
      *(v4u*)&sB2[nxt][brow + 64][bko] = r21;
    }
    __syncthreads();
  }

  #pragma unroll
  for (int s = 0; s < 2; ++s) {
    const int ncol = n_base + wn * 32 + s * 16 + l16;
    const float bias1 = b1[e * DH + ncol];
    const float bias2 = b2[e * DH + ncol];
    #pragma unroll
    for (int r = 0; r < 8; ++r) {
      const int slot = m_base + wm * 16 + hh * 8 + r;   // C/D: M = r + 8*half
      const float hv = hacc[s][r] + bias1;
      const float gv = gacc[s][r] + bias2;
      const float sil = gv / (1.0f + __expf(-gv));
      act[((long)e * TKN + slot) * DH + ncol] = f2bf(hv * sil);
    }
  }
}

// ------------------------------------------------ FFN2: out += gate * (act @ Wo + bo)
// Wot is [e][n(DM)][k(DH)] bf16. Block tile M=32 x N=128.
__global__ __launch_bounds__(256) void moe_ffn2(
    const u16* __restrict__ act, const u16* __restrict__ Wot,
    const float* __restrict__ bo, const int* __restrict__ counts,
    const int* __restrict__ idxl, const float* __restrict__ gatel,
    float* __restrict__ out)
{
  __shared__ u16 sA[2][32][40];
  __shared__ u16 sB[2][128][40];

  const int e = blockIdx.z;
  const int cnt = counts[e];
  const int m_base = blockIdx.y * 32;
  if (m_base >= cnt) return;
  const int n_base = blockIdx.x * 128;
  const int tid  = threadIdx.x;
  const int lane = tid & 31;
  const int w    = tid >> 5;
  const int wm   = w >> 2;
  const int wn   = w & 3;
  const int hh   = lane >> 4;
  const int l16  = lane & 15;

  const int arow = tid >> 2;
  const int ak   = (tid & 3) * 8;
  const u16* pA = act + ((long)e * TKN + m_base + (tid < 128 ? arow : 0)) * DH + ak;

  const int brow = tid >> 2;
  const int bko  = (tid & 3) * 8;
  const u16* pBa = Wot + (long)e * DM * DH + (long)(n_base + brow) * DH + bko;
  const u16* pBb = pBa + (long)64 * DH;

  if (tid < 128) *(v4u*)&sA[0][arow][ak] = *(const v4u*)pA;
  *(v4u*)&sB[0][brow][bko]      = *(const v4u*)pBa;
  *(v4u*)&sB[0][brow + 64][bko] = *(const v4u*)pBb;
  __syncthreads();

  v8f yacc[2] = {{}, {}};

  const int NT = DH / 32;
  for (int kt = 0; kt < NT; ++kt) {
    const int cur = kt & 1;
    const int kn  = (kt + 1) * 32;
    const bool more = (kt + 1 < NT);
    v4u ra = {}, rb0 = {}, rb1 = {};
    if (more) {
      if (tid < 128) ra = *(const v4u*)(pA + kn);
      rb0 = *(const v4u*)(pBa + kn);
      rb1 = *(const v4u*)(pBb + kn);
    }
    FragBF fa;
    fa.q[0] = *(const v4u*)&sA[cur][wm * 16 + l16][16 * hh];
    fa.q[1] = *(const v4u*)&sA[cur][wm * 16 + l16][16 * hh + 8];
    #pragma unroll
    for (int s = 0; s < 2; ++s) {
      const int nl = wn * 32 + s * 16 + l16;
      FragBF fb;
      fb.q[0] = *(const v4u*)&sB[cur][nl][16 * hh];
      fb.q[1] = *(const v4u*)&sB[cur][nl][16 * hh + 8];
      yacc[s] = bfwmma(fa, fb, yacc[s]);
    }
    if (more) {
      const int nxt = cur ^ 1;
      if (tid < 128) *(v4u*)&sA[nxt][arow][ak] = ra;
      *(v4u*)&sB[nxt][brow][bko]      = rb0;
      *(v4u*)&sB[nxt][brow + 64][bko] = rb1;
    }
    __syncthreads();
  }

  #pragma unroll
  for (int s = 0; s < 2; ++s) {
    const int ncol = n_base + wn * 32 + s * 16 + l16;
    const float biaso = bo[e * DM + ncol];
    #pragma unroll
    for (int r = 0; r < 8; ++r) {
      const int slot = m_base + wm * 16 + hh * 8 + r;
      const int   tok  = idxl[e * TKN + slot];
      const float gate = gatel[e * TKN + slot];     // 0 for padded slots
      const float v = (yacc[s][r] + biaso) * gate;
      atomAddF32(&out[(long)tok * DM + ncol], v);
    }
  }
}

// ------------------------------------------------ host
extern "C" void kernel_launch(void* const* d_in, const int* in_sizes, int n_in,
                              void* d_out, int out_size, void* d_ws, size_t ws_size,
                              hipStream_t stream) {
  (void)in_sizes; (void)n_in; (void)out_size; (void)ws_size;
  const float* x  = (const float*)d_in[0];
  const float* W1 = (const float*)d_in[1];
  const float* b1 = (const float*)d_in[2];
  const float* W2 = (const float*)d_in[3];
  const float* b2 = (const float*)d_in[4];
  const float* Wo = (const float*)d_in[5];
  const float* bo = (const float*)d_in[6];
  const float* Wr = (const float*)d_in[7];
  const float* br = (const float*)d_in[8];
  float* out = (float*)d_out;

  size_t off = 0;
  auto alloc = [&](size_t bytes) -> void* {
    void* p = (char*)d_ws + off;
    off = (off + bytes + 255) & ~(size_t)255;
    return p;
  };
  u16*   xb    = (u16*)  alloc((size_t)TKN * DM * 2);
  u16*   W1t   = (u16*)  alloc((size_t)NE * DM * DH * 2);
  u16*   W2t   = (u16*)  alloc((size_t)NE * DM * DH * 2);
  u16*   Wot   = (u16*)  alloc((size_t)NE * DH * DM * 2);
  u16*   actb  = (u16*)  alloc((size_t)NE * TKN * DH * 2);
  int*   idxl  = (int*)  alloc((size_t)NE * TKN * 4);
  float* gatel = (float*)alloc((size_t)NE * TKN * 4);
  int*   counts= (int*)  alloc((size_t)NE * 4);
  float* psum  = (float*)alloc((size_t)NE * 4);

  moe_prep<<<512, 256, 0, stream>>>(x, xb, idxl, gatel, counts, psum, out);
  moe_transpose_bf16<<<dim3(DH / 32, DM / 32, NE), 256, 0, stream>>>(W1, W1t, DM, DH);
  moe_transpose_bf16<<<dim3(DH / 32, DM / 32, NE), 256, 0, stream>>>(W2, W2t, DM, DH);
  moe_transpose_bf16<<<dim3(DM / 32, DH / 32, NE), 256, 0, stream>>>(Wo, Wot, DH, DM);
  moe_router<<<TKN / 256, 256, 0, stream>>>(x, Wr, br, idxl, gatel, counts, psum);
  moe_loss<<<1, 32, 0, stream>>>(psum, out + (size_t)TKN * DM);
  moe_ffn1<<<dim3(DH / 128, TKN / 32, NE), 256, 0, stream>>>(xb, W1t, W2t, b1, b2,
                                                             counts, idxl, actb);
  moe_ffn2<<<dim3(DM / 128, TKN / 32, NE), 256, 0, stream>>>(actb, Wot, bo,
                                                             counts, idxl, gatel, out);
}
